// CompositionalEmbedding_18313740550722
// MI455X (gfx1250) — matrix operand order repaired and checked by reading, not compiled, not run on hardware
//
#include <hip/hip_runtime.h>

typedef __attribute__((ext_vector_type(16))) _Float16 v16h;
typedef __attribute__((ext_vector_type(8)))  float    v8f;

constexpr int E     = 128;   // component dim
constexpr int TOK   = 5;     // tokens per item
constexpr int ITEMS = 16;    // items per workgroup
constexpr int ROWS  = ITEMS * TOK;   // 80 token rows
constexpr int NH    = 4;     // heads
constexpr int DH    = 32;    // head dim
constexpr int OUTD  = 768;   // output embedding dim

constexpr int CSTR = E + 8;       // 136 halves (comp / attn-out stride) -> 272B rows
constexpr int QSTR = 3 * E + 8;   // 392 halves (qkv stride) -> 784B rows
constexpr int MSTR = E + 8;       // composed stride

// f16 weight layout inside d_ws (element offsets)
constexpr int WQKV_OFF = 0;                   // [384][128]
constexpr int WOP_OFF  = 3 * E * E;           // 49152, [128][128]
constexpr int WOUT_OFF = WOP_OFF + E * E;     // 65536, [768][128]
constexpr int WTOTAL   = WOUT_OFF + OUTD * E; // 163840 halves

union FragU  { v16h v; uint4 q[2]; };
union Pack8  { uint4 q;    _Float16 h[8];  };
union Pack32 { uint4 q[4]; _Float16 h[32]; };

// A fragment 16x32 f16 from LDS (row-major, stride in halves).
// lane<16: M=lane, K = [k0..k0+8) and [k0+16..k0+24)
// lane>=16: M=lane-16, K = [k0+8..k0+16) and [k0+24..k0+32)
__device__ __forceinline__ v16h load_frag_a(const _Float16* base, int stride, int m0, int k0) {
  const int lane = threadIdx.x & 31;
  const int l = lane & 15, hi = lane >> 4;
  const _Float16* p = base + (m0 + l) * stride + k0 + hi * 8;
  FragU f;
  f.q[0] = *reinterpret_cast<const uint4*>(p);
  f.q[1] = *reinterpret_cast<const uint4*>(p + 16);
  return f.v;
}

// B fragment 32x16 f16 from global weights stored [n][K=128] row-major.
// Column n = lane%16; lanes 0-15 hold K=[k0..k0+8) and [k0+16..k0+24), lanes 16-31 K=[k0+8..k0+16) and [k0+24..k0+32).
__device__ __forceinline__ v16h load_frag_b(const _Float16* w, int n0, int k0) {
  const int lane = threadIdx.x & 31;
  const int l = lane & 15, hi = lane >> 4;
  const _Float16* p = w + (n0 + l) * E + k0 + hi * 8;
  FragU f;
  f.q[0] = *reinterpret_cast<const uint4*>(p);
  f.q[1] = *reinterpret_cast<const uint4*>(p + 16);
  return f.v;
}

__device__ __forceinline__ void load_row32(const _Float16* p, float* f) {
  Pack32 pk;
  pk.q[0] = *reinterpret_cast<const uint4*>(p);
  pk.q[1] = *reinterpret_cast<const uint4*>(p + 8);
  pk.q[2] = *reinterpret_cast<const uint4*>(p + 16);
  pk.q[3] = *reinterpret_cast<const uint4*>(p + 24);
  #pragma unroll
  for (int d = 0; d < 32; ++d) f[d] = (float)pk.h[d];
}

__global__ void ce_prep_weights(const float* __restrict__ wqkv,
                                const float* __restrict__ wop,
                                const float* __restrict__ wout,
                                _Float16* __restrict__ dst) {
  int i = blockIdx.x * blockDim.x + threadIdx.x;
  if (i >= WTOTAL) return;
  float v;
  if (i < WOP_OFF)       v = wqkv[i];
  else if (i < WOUT_OFF) v = wop[i - WOP_OFF];
  else                   v = wout[i - WOUT_OFF];
  dst[i] = (_Float16)v;
}

__global__ __launch_bounds__(128)
void ce_main(const int* __restrict__ device_ids, const int* __restrict__ pseudo_ids,
             const int* __restrict__ attr_ids, const int* __restrict__ unit_ids,
             const float* __restrict__ values, const int* __restrict__ mask,
             const float* __restrict__ dev_table, const float* __restrict__ pseudo_table,
             const float* __restrict__ attr_table, const float* __restrict__ unit_table,
             const float* __restrict__ val_w, const float* __restrict__ val_b,
             const float* __restrict__ in_proj_b, const float* __restrict__ out_proj_b,
             const float* __restrict__ out_b,
             const _Float16* __restrict__ wf16,
             float* __restrict__ out, int n) {
  extern __shared__ _Float16 smem[];
  _Float16* sA = smem;                    // ROWS*CSTR : comp (ph0/1), attn-out (ph2/3)
  _Float16* sB = smem + ROWS * CSTR;      // ROWS*QSTR : qkv (ph1/2), out_proj result (ph3/4)
  _Float16* sC = sB + ROWS * QSTR;        // ITEMS*MSTR : composed mean

  const int tid   = threadIdx.x;
  const int wave  = tid >> 5;
  const int lane  = tid & 31;
  const int l     = lane & 15;
  const int hi    = lane >> 4;
  const int item0 = blockIdx.x * ITEMS;

  // Warm L2 for the f16 weight stream (global_prefetch_b8).
  __builtin_prefetch(wf16 + ((size_t)tid << 8), 0, 1);
  __builtin_prefetch(wf16 + WOUT_OFF + ((size_t)tid << 8), 0, 1);

  // ---------- Phase 0: gather + mask -> comp f16 [80 x 128] ----------
  for (int t = tid; t < ROWS * (E / 8); t += blockDim.x) {
    const int row = t >> 4;
    const int c0  = (t & 15) * 8;
    const int it  = row / TOK;
    const int tok = row - it * TOK;
    int g = item0 + it;
    const bool ok = (g < n);
    if (!ok) g = 0;
    const float sc = ok ? (float)mask[g * TOK + tok] : 0.f;
    float v[8];
    if (tok == 3) {
      const float x = values[g];
      #pragma unroll
      for (int j = 0; j < 8; ++j) v[j] = x * val_w[c0 + j] + val_b[c0 + j];
    } else {
      const float* src;
      if      (tok == 0) src = dev_table    + device_ids[g] * E;
      else if (tok == 1) src = pseudo_table + pseudo_ids[g] * E;
      else if (tok == 2) src = attr_table   + attr_ids[g]   * E;
      else               src = unit_table   + unit_ids[g]   * E;
      #pragma unroll
      for (int j = 0; j < 8; ++j) v[j] = src[c0 + j];
    }
    Pack8 pk;
    #pragma unroll
    for (int j = 0; j < 8; ++j) pk.h[j] = (_Float16)(v[j] * sc);
    *reinterpret_cast<uint4*>(&sA[row * CSTR + c0]) = pk.q;  // one ds_store_b128
  }
  __syncthreads();

  // ---------- Phase 1: QKV = comp @ Wqkv^T + b  -> sB [80 x 384] ----------
  {
    const _Float16* wqkv = wf16 + WQKV_OFF;
    for (int t = wave; t < (ROWS / 16) * (3 * E / 16); t += 4) {
      const int mt = t / (3 * E / 16);
      const int nt = t % (3 * E / 16);
      const int col = nt * 16 + l;
      const float bias = in_proj_b[col];
      v8f acc = {bias, bias, bias, bias, bias, bias, bias, bias}; // bias folded into C
      #pragma unroll
      for (int kt = 0; kt < E / 32; ++kt) {
        v16h a = load_frag_a(sA, CSTR, mt * 16, kt * 32);
        v16h b = load_frag_b(wqkv, nt * 16, kt * 32);
        acc = __builtin_amdgcn_wmma_f32_16x16x32_f16(false, a, false, b,
                                                     (short)0, acc, false, false);
      }
      #pragma unroll
      for (int j = 0; j < 8; ++j) {
        const int row = mt * 16 + hi * 8 + j;
        sB[row * QSTR + col] = (_Float16)acc[j];
      }
    }
  }
  __syncthreads();

  // ---------- Phase 2: 5-token MHA per (item, head) -> sA [80 x 128] ----------
  if (tid < ITEMS * NH) {
    const int it = tid >> 2;
    const int hd = tid & 3;
    const int qc = hd * DH, kc = E + hd * DH, vc = 2 * E + hd * DH;
    const _Float16* rows = sB + (it * TOK) * QSTR;
    for (int qi = 0; qi < TOK; ++qi) {
      float qr[DH];
      load_row32(rows + qi * QSTR + qc, qr);
      float s[TOK];
      float mx = -1e30f;
      for (int ki = 0; ki < TOK; ++ki) {
        float kr[DH];
        load_row32(rows + ki * QSTR + kc, kr);
        float dot = 0.f;
        #pragma unroll
        for (int d = 0; d < DH; ++d) dot += qr[d] * kr[d];
        s[ki] = dot * 0.17677669529663687f; // 1/sqrt(32)
        mx = fmaxf(mx, s[ki]);
      }
      float sum = 0.f;
      #pragma unroll
      for (int ki = 0; ki < TOK; ++ki) { s[ki] = __expf(s[ki] - mx); sum += s[ki]; }
      const float inv = 1.f / sum;
      float o[DH];
      #pragma unroll
      for (int d = 0; d < DH; ++d) o[d] = 0.f;
      for (int ki = 0; ki < TOK; ++ki) {
        const float w = s[ki] * inv;
        float vr[DH];
        load_row32(rows + ki * QSTR + vc, vr);
        #pragma unroll
        for (int d = 0; d < DH; ++d) o[d] += w * vr[d];
      }
      Pack32 pk;
      #pragma unroll
      for (int d = 0; d < DH; ++d) pk.h[d] = (_Float16)o[d];
      _Float16* dst = &sA[(it * TOK + qi) * CSTR + qc];
      *reinterpret_cast<uint4*>(dst)      = pk.q[0];
      *reinterpret_cast<uint4*>(dst + 8)  = pk.q[1];
      *reinterpret_cast<uint4*>(dst + 16) = pk.q[2];
      *reinterpret_cast<uint4*>(dst + 24) = pk.q[3];
    }
  }
  __syncthreads();

  // ---------- Phase 3: out_proj GEMM [80x128]x[128x128] + bias -> sB [80 x 136] ----------
  {
    const _Float16* wop = wf16 + WOP_OFF;
    for (int t = wave; t < (ROWS / 16) * (E / 16); t += 4) {
      const int mt = t / (E / 16);
      const int nt = t % (E / 16);
      const int col = nt * 16 + l;
      const float bias = out_proj_b[col];
      v8f acc = {bias, bias, bias, bias, bias, bias, bias, bias};
      #pragma unroll
      for (int kt = 0; kt < E / 32; ++kt) {
        v16h a = load_frag_a(sA, CSTR, mt * 16, kt * 32);
        v16h b = load_frag_b(wop, nt * 16, kt * 32);
        acc = __builtin_amdgcn_wmma_f32_16x16x32_f16(false, a, false, b,
                                                     (short)0, acc, false, false);
      }
      #pragma unroll
      for (int j = 0; j < 8; ++j) {
        const int row = mt * 16 + hi * 8 + j;
        sB[row * CSTR + col] = (_Float16)acc[j];
      }
    }
  }
  __syncthreads();

  // ---------- Phase 4: mean over 5 tokens -> sC [16 x 128] ----------
  for (int t = tid; t < ITEMS * (E / 8); t += blockDim.x) {  // 256 8-wide tasks
    const int it = t >> 4;
    const int c0 = (t & 15) * 8;
    float s[8];
    #pragma unroll
    for (int j = 0; j < 8; ++j) s[j] = 0.f;
    #pragma unroll
    for (int k = 0; k < TOK; ++k) {
      Pack8 pk;
      pk.q = *reinterpret_cast<const uint4*>(&sB[(it * TOK + k) * CSTR + c0]);
      #pragma unroll
      for (int j = 0; j < 8; ++j) s[j] += (float)pk.h[j];
    }
    Pack8 o;
    #pragma unroll
    for (int j = 0; j < 8; ++j) o.h[j] = (_Float16)(s[j] * 0.2f);
    *reinterpret_cast<uint4*>(&sC[it * MSTR + c0]) = o.q;
  }
  __syncthreads();

  // ---------- Phase 5: final GEMM [16x128]x[128x768] + bias -> d_out ----------
  {
    const _Float16* wout = wf16 + WOUT_OFF;
    for (int nt = wave; nt < OUTD / 16; nt += 4) {
      const int col = nt * 16 + l;
      const float bias = out_b[col];
      v8f acc = {bias, bias, bias, bias, bias, bias, bias, bias};
      #pragma unroll
      for (int kt = 0; kt < E / 32; ++kt) {
        v16h a = load_frag_a(sC, MSTR, 0, kt * 32);
        v16h b = load_frag_b(wout, nt * 16, kt * 32);
        acc = __builtin_amdgcn_wmma_f32_16x16x32_f16(false, a, false, b,
                                                     (short)0, acc, false, false);
      }
      #pragma unroll
      for (int j = 0; j < 8; ++j) {
        const int g = item0 + hi * 8 + j;
        if (g < n) out[g * OUTD + col] = acc[j];
      }
    }
  }
}

extern "C" void kernel_launch(void* const* d_in, const int* in_sizes, int n_in,
                              void* d_out, int out_size, void* d_ws, size_t ws_size,
                              hipStream_t stream) {
  const int*   device_ids   = (const int*)d_in[0];
  const int*   pseudo_ids   = (const int*)d_in[1];
  const int*   attr_ids     = (const int*)d_in[2];
  const int*   unit_ids     = (const int*)d_in[3];
  const float* values       = (const float*)d_in[4];
  const int*   mask         = (const int*)d_in[5];
  const float* dev_table    = (const float*)d_in[6];
  const float* pseudo_table = (const float*)d_in[7];
  const float* attr_table   = (const float*)d_in[8];
  const float* unit_table   = (const float*)d_in[9];
  const float* val_w        = (const float*)d_in[10];
  const float* val_b        = (const float*)d_in[11];
  const float* in_proj_w    = (const float*)d_in[12];
  const float* in_proj_b    = (const float*)d_in[13];
  const float* out_proj_w   = (const float*)d_in[14];
  const float* out_proj_b   = (const float*)d_in[15];
  const float* out_w        = (const float*)d_in[16];
  const float* out_b        = (const float*)d_in[17];

  const int n = in_sizes[0];  // N items
  _Float16* wf16 = (_Float16*)d_ws;

  ce_prep_weights<<<(WTOTAL + 255) / 256, 256, 0, stream>>>(in_proj_w, out_proj_w, out_w, wf16);

  constexpr size_t SMEM =
      (size_t)(ROWS * CSTR + ROWS * QSTR + ITEMS * MSTR) * sizeof(_Float16); // ~87 KB
  const int blocks = (n + ITEMS - 1) / ITEMS;
  ce_main<<<blocks, 128, SMEM, stream>>>(
      device_ids, pseudo_ids, attr_ids, unit_ids, values, mask,
      dev_table, pseudo_table, attr_table, unit_table,
      val_w, val_b, in_proj_b, out_proj_b, out_b,
      wf16, (float*)d_out, n);
}
